// GL_SAGEConv_9l_128h_44753559224360
// MI455X (gfx1250) — matrix operand order repaired and checked by reading, not compiled, not run on hardware
//
#include <hip/hip_runtime.h>
#include <hip/hip_bf16.h>

typedef __attribute__((ext_vector_type(16))) _Float16 v16h;
typedef __attribute__((ext_vector_type(8)))  float    v8f;

#define N_NODES 50000
#define N_EDGES 600000
#define D 128
#define N_CLASSES 10
#define ROW_BLOCKS 3125   // 50000 / 16

// ---------------- utility kernels ----------------

__global__ void zero_f32(float* __restrict__ p, int n) {
  int i = blockIdx.x * blockDim.x + threadIdx.x;
  if (i < n) p[i] = 0.0f;
}

__global__ void deg_kernel(const long long* __restrict__ ei, float* __restrict__ deg) {
  int e = blockIdx.x * blockDim.x + threadIdx.x;
  if (e < N_EDGES) {
    int dst = (int)ei[N_EDGES + e];
    atomicAdd(&deg[dst], 1.0f);
  }
}

__global__ void invdeg_kernel(float* __restrict__ deg) {
  int i = blockIdx.x * blockDim.x + threadIdx.x;
  if (i < N_NODES) deg[i] = 1.0f / fmaxf(deg[i], 1.0f);
}

// Gather h[src] row, atomically accumulate into agg[dst] row.
// 32 threads (one wave) per edge, 4 floats per thread -> fully coalesced 512B
// gather + scatter. h (25.6MB) and agg stay resident in the 192MB L2.
__global__ void scatter_kernel(const float* __restrict__ h,
                               const long long* __restrict__ ei,
                               float* __restrict__ agg) {
  int tid = blockIdx.x * blockDim.x + threadIdx.x;
  int e   = tid >> 5;
  int seg = tid & 31;
  if (e >= N_EDGES) return;
  int src = (int)ei[e];
  int dst = (int)ei[N_EDGES + e];
  const float4 v = *(const float4*)(h + src * D + seg * 4);
  float* a = agg + dst * D + seg * 4;
  atomicAdd(a + 0, v.x);
  atomicAdd(a + 1, v.y);
  atomicAdd(a + 2, v.z);
  atomicAdd(a + 3, v.w);
}

// Pre-pack a f32 weight matrix W[K=128][N=ncols] into f16 WMMA B-fragment
// order: pk[((tile*4 + chunk)*32 + lane)*16 + e] = W[K][N], with
//   K = chunk*32 + (lane/16)*8 + (e<8 ? e : 8+e),  N = tile*16 + lane%16  (zero-pad N>=ncols)
// so each lane's 16-half fragment is contiguous (two b128 loads).
__global__ void pack_w(const float* __restrict__ W, _Float16* __restrict__ pk,
                       int ncols, int ntiles) {
  int idx = blockIdx.x * blockDim.x + threadIdx.x;
  int total = ntiles * 4 * 32 * 16;
  if (idx >= total) return;
  int e    = idx & 15;
  int lane = (idx >> 4) & 31;
  int c    = (idx >> 9) & 3;
  int tile = idx >> 11;
  int K = c * 32 + (lane >> 4) * 8 + (e < 8 ? e : 8 + e);
  int N = tile * 16 + (lane & 15);
  float v = (N < ncols) ? W[K * ncols + N] : 0.0f;
  pk[idx] = (_Float16)v;
}

// ---------------- fused SAGE layer GEMM ----------------
// out[m][n] = act( (agg[m]*invd[m]) @ Wl + h[m] @ Wr + bias )  for 16 rows/block.
// 8 waves: wave w computes N-tile w (cols 16w..16w+15) with 8 chained
// v_wmma_f32_16x16x32_f16 into one f32 accumulator.
__global__ __launch_bounds__(256)
void sage_gemm(const float* __restrict__ agg, const float* __restrict__ invd,
               const float* __restrict__ h,
               const _Float16* __restrict__ Wl_pk, const _Float16* __restrict__ Wr_pk,
               const float* __restrict__ bias, float* __restrict__ out,
               int ncols, int ntiles, int do_elu) {
  // padded stride 136 halves (272B = 68 words) -> 16 rows land on distinct banks
  __shared__ __align__(16) _Float16 sAgg[16][136];
  __shared__ __align__(16) _Float16 sH[16][136];

  const int tid = threadIdx.x;
  const int m0  = blockIdx.x * 16;

  // cooperative staging: f32 -> f16, inv_deg fused into agg tile
  for (int i = tid; i < 16 * D; i += 256) {
    int r = i >> 7, col = i & 127;
    int g = (m0 + r) * D + col;
    sAgg[r][col] = (_Float16)(agg[g] * invd[m0 + r]);
    sH[r][col]   = (_Float16)h[g];
  }
  __syncthreads();

  const int wave = tid >> 5;
  const int lane = tid & 31;
  if (wave >= ntiles) return;

  const int row  = lane & 15;   // A: M = lane%16 ; C/D: N = lane%16
  const int half = lane >> 4;

  v8f acc = {};
  const _Float16* pkL = Wl_pk + ((size_t)(wave * 4) * 32 + lane) * 16;
  const _Float16* pkR = Wr_pk + ((size_t)(wave * 4) * 32 + lane) * 16;

#pragma unroll
  for (int c = 0; c < 4; ++c) {
    const int kb = c * 32 + half * 8;
    union { v16h v; float4 f4[2]; } aAgg, aH;
    aAgg.f4[0] = *(const float4*)&sAgg[row][kb];        // K = kb .. kb+7
    aAgg.f4[1] = *(const float4*)&sAgg[row][kb + 16];   // K = kb+16 .. kb+23
    aH.f4[0]   = *(const float4*)&sH[row][kb];
    aH.f4[1]   = *(const float4*)&sH[row][kb + 16];
    v16h bL = *(const v16h*)(pkL + c * 32 * 16);
    v16h bR = *(const v16h*)(pkR + c * 32 * 16);
    acc = __builtin_amdgcn_wmma_f32_16x16x32_f16(false, aAgg.v, false, bL,
                                                 (short)0, acc, false, false);
    acc = __builtin_amdgcn_wmma_f32_16x16x32_f16(false, aH.v, false, bR,
                                                 (short)0, acc, false, false);
  }

  const int col = wave * 16 + row;      // output column
  if (col < ncols) {
    const float b = bias[col];
#pragma unroll
    for (int r = 0; r < 8; ++r) {
      int m = m0 + r + half * 8;        // C layout: M = r + (lane/16)*8
      float v = acc[r] + b;
      if (do_elu) v = (v > 0.0f) ? v : (__expf(v) - 1.0f);
      out[m * ncols + col] = v;
    }
  }
}

// ---------------- host-side orchestration ----------------

extern "C" void kernel_launch(void* const* d_in, const int* in_sizes, int n_in,
                              void* d_out, int out_size, void* d_ws, size_t ws_size,
                              hipStream_t stream) {
  const float*     x      = (const float*)d_in[0];
  const long long* ei     = (const long long*)d_in[1];
  const float*     Wl     = (const float*)d_in[2];  // [8][128][128]
  const float*     bl     = (const float*)d_in[3];  // [8][128]
  const float*     Wr     = (const float*)d_in[4];  // [8][128][128]
  const float*     Wl_out = (const float*)d_in[5];  // [128][10]
  const float*     bl_out = (const float*)d_in[6];  // [10]
  const float*     Wr_out = (const float*)d_in[7];  // [128][10]

  char* ws = (char*)d_ws;
  size_t off = 0;
  auto alloc = [&](size_t bytes) -> void* {
    void* p = ws + off;
    off += bytes;
    off = (off + 255) & ~(size_t)255;
    return p;
  };

  const size_t featBytes = (size_t)N_NODES * D * sizeof(float);
  float* hA   = (float*)alloc(featBytes);
  float* hB   = (float*)alloc(featBytes);
  float* agg  = (float*)alloc(featBytes);
  float* invd = (float*)alloc((size_t)N_NODES * sizeof(float));
  const int HID_PK = 8 * 4 * 32 * 16;   // halves per packed 128x128 matrix
  const int OUT_PK = 1 * 4 * 32 * 16;   // halves per packed 128x16 matrix
  _Float16* pkWl  = (_Float16*)alloc((size_t)8 * HID_PK * sizeof(_Float16));
  _Float16* pkWr  = (_Float16*)alloc((size_t)8 * HID_PK * sizeof(_Float16));
  _Float16* pkWlo = (_Float16*)alloc((size_t)OUT_PK * sizeof(_Float16));
  _Float16* pkWro = (_Float16*)alloc((size_t)OUT_PK * sizeof(_Float16));

  // pack all weights to WMMA fragment layout (one-time, tiny)
  for (int i = 0; i < 8; ++i) {
    pack_w<<<(HID_PK + 255) / 256, 256, 0, stream>>>(Wl + (size_t)i * D * D,
                                                     pkWl + (size_t)i * HID_PK, D, 8);
    pack_w<<<(HID_PK + 255) / 256, 256, 0, stream>>>(Wr + (size_t)i * D * D,
                                                     pkWr + (size_t)i * HID_PK, D, 8);
  }
  pack_w<<<(OUT_PK + 255) / 256, 256, 0, stream>>>(Wl_out, pkWlo, N_CLASSES, 1);
  pack_w<<<(OUT_PK + 255) / 256, 256, 0, stream>>>(Wr_out, pkWro, N_CLASSES, 1);

  // degrees -> inv_deg
  zero_f32<<<(N_NODES + 255) / 256, 256, 0, stream>>>(invd, N_NODES);
  deg_kernel<<<(N_EDGES + 255) / 256, 256, 0, stream>>>(ei, invd);
  invdeg_kernel<<<(N_NODES + 255) / 256, 256, 0, stream>>>(invd);

  const int nAgg = N_NODES * D;
  const int scatterThreads = N_EDGES * 32;

  const float* cur = x;
  float* bufs[2] = {hA, hB};
  for (int i = 0; i < 8; ++i) {
    zero_f32<<<(nAgg + 255) / 256, 256, 0, stream>>>(agg, nAgg);
    scatter_kernel<<<(scatterThreads + 255) / 256, 256, 0, stream>>>(cur, ei, agg);
    sage_gemm<<<ROW_BLOCKS, 256, 0, stream>>>(agg, invd, cur,
                                              pkWl + (size_t)i * HID_PK,
                                              pkWr + (size_t)i * HID_PK,
                                              bl + (size_t)i * D,
                                              bufs[i & 1], D, 8, 1);
    cur = bufs[i & 1];
  }

  // output layer (no activation), cols padded 10 -> 16 in packed weights
  zero_f32<<<(nAgg + 255) / 256, 256, 0, stream>>>(agg, nAgg);
  scatter_kernel<<<(scatterThreads + 255) / 256, 256, 0, stream>>>(cur, ei, agg);
  sage_gemm<<<ROW_BLOCKS, 256, 0, stream>>>(agg, invd, cur, pkWlo, pkWro, bl_out,
                                            (float*)d_out, N_CLASSES, 1, 0);
}
